// CausalSelfAttention_27565100106218
// MI455X (gfx1250) — matrix operand order repaired and checked by reading, not compiled, not run on hardware
//
#include <hip/hip_runtime.h>
#include <hip/hip_bf16.h>
#include <cstdint>
#include <cstddef>

typedef _Float16 h16;
typedef __attribute__((ext_vector_type(8)))  _Float16 v8h;
typedef __attribute__((ext_vector_type(16))) _Float16 v16h;
typedef __attribute__((ext_vector_type(8)))  float    v8f;

#define B_  4
#define T_  2048
#define C_  1024
#define NH_ 16
#define HS_ 64

union V16u { v16h v; v8h h[2]; };
static __device__ inline v16h make16(v8h lo, v8h hi) { V16u u; u.h[0] = lo; u.h[1] = hi; return u.v; }

// 32-bit LDS byte offset of a generic pointer to __shared__ memory
static __device__ inline unsigned lds_off(const void* p) {
  return (unsigned)(size_t)(__attribute__((address_space(3))) const char*)p;
}

// async global->LDS copy of 16 bytes (ASYNCcnt-tracked, no VGPR staging)
static __device__ inline void async_cp16(unsigned lds_byte, const void* gptr) {
  asm volatile("global_load_async_to_lds_b128 %0, %1, off"
               :: "v"(lds_byte), "v"((unsigned long long)(size_t)gptr) : "memory");
}
static __device__ inline void async_cp32(unsigned lds_byte, const void* gptr) {
  unsigned long long ga = (unsigned long long)(size_t)gptr;
  asm volatile("global_load_async_to_lds_b128 %0, %1, off"           :: "v"(lds_byte), "v"(ga) : "memory");
  asm volatile("global_load_async_to_lds_b128 %0, %1, off offset:16" :: "v"(lds_byte), "v"(ga) : "memory");
}
static __device__ inline void wait_async0() {
  asm volatile("s_wait_asynccnt 0x0" ::: "memory");
}

// ---------------------------------------------------------------- f32 -> f16
__global__ void cvt_f32_f16(const float* __restrict__ src, h16* __restrict__ dst, int n) {
  int i = blockIdx.x * blockDim.x + threadIdx.x;
  int stride = gridDim.x * blockDim.x;
  for (; i < n; i += stride) dst[i] = (h16)src[i];
}

// ---------------------------------------------------------------- RoPE (in-place on f16 qkv)
__global__ void rope_kernel(h16* __restrict__ qkv) {
  int idx = blockIdx.x * blockDim.x + threadIdx.x;      // (b*T + t)*NH + h
  if (idx >= B_ * T_ * NH_) return;
  int h  = idx % NH_;
  int bt = idx / NH_;
  int t  = bt % T_;
  size_t base = (size_t)bt * (3 * C_) + (size_t)h * HS_;
  h16* q = qkv + base;
  h16* k = qkv + base + C_;
  float qe[32], qo[32], ke[32], ko[32];
  for (int i = 0; i < 32; i++) {
    qe[i] = (float)q[2*i]; qo[i] = (float)q[2*i+1];
    ke[i] = (float)k[2*i]; ko[i] = (float)k[2*i+1];
  }
  for (int i = 0; i < 32; i++) {
    // inv_freq = 1000^(-2i/64) = exp(-ln(1000)*2i/64)
    float f = __expf(-(float)(2*i) * (6.907755278982137f / 64.0f));
    float ang = (float)t * f;
    float s, c;
    __sincosf(ang, &s, &c);
    q[i]      = (h16)(qe[i]*c - qo[i]*s);
    q[32 + i] = (h16)(qo[i]*c + qe[i]*s);
    k[i]      = (h16)(ke[i]*c - ko[i]*s);
    k[32 + i] = (h16)(ko[i]*c + ke[i]*s);
  }
}

// ---------------------------------------------------------------- tiled WMMA GEMM: out = A(MxK) * Bw(NxK)^T + bias
// A, Bw row-major f16 (K contiguous). 128x128 workgroup tile, 8 waves (4M x 2N),
// each wave: 2x4 WMMA tiles of 16x16, K-step 32. Tiles staged with async-to-LDS.
template <bool OUT_F32>
__global__ __launch_bounds__(256)
void gemm_bias(const h16* __restrict__ A, const h16* __restrict__ Bw,
               const float* __restrict__ bias, void* __restrict__ outp,
               int M, int N, int K) {
  __shared__ h16 As[128][40];
  __shared__ h16 Bs[128][40];
  const int tid  = threadIdx.x;
  const int lane = tid & 31, w = tid >> 5;
  const int g = lane >> 4, n16 = lane & 15;
  const int wm = w & 3, wn = w >> 2;           // 4 waves along M, 2 along N
  const int m0 = blockIdx.y * 128, n0 = blockIdx.x * 128;

  v8f acc[2][4];
  for (int i = 0; i < 2; i++) for (int j = 0; j < 4; j++) acc[i][j] = {};

  const int srow = tid >> 1, scol = (tid & 1) * 16;  // each thread stages 16 halfs/row of A and B
  const unsigned ldsA = lds_off(&As[srow][scol]);
  const unsigned ldsB = lds_off(&Bs[srow][scol]);

  for (int k0 = 0; k0 < K; k0 += 32) {
    const h16* ag = A  + (size_t)(m0 + srow) * K + k0 + scol;
    const h16* bg = Bw + (size_t)(n0 + srow) * K + k0 + scol;
    async_cp32(ldsA, ag);                      // global_load_async_to_lds_b128 x2
    async_cp32(ldsB, bg);
    if (k0 + 32 < K) {                         // warm L2 for the next K tile
      __builtin_prefetch(ag + 32, 0, 1);
      __builtin_prefetch(bg + 32, 0, 1);
    }
    wait_async0();
    __syncthreads();

    v16h af[2], bf[4];
    for (int mt = 0; mt < 2; mt++) {
      int r = wm * 32 + mt * 16 + n16;         // A frag: row per lane, K = 8g+e | 16+8g+(e-8)
      af[mt] = make16(*(const v8h*)&As[r][8 * g], *(const v8h*)&As[r][16 + 8 * g]);
    }
    for (int nt = 0; nt < 4; nt++) {
      int r = wn * 64 + nt * 16 + n16;         // B frag: N per lane, K = 8g+e | 16+8g+(e-8) (Bs is [N][K])
      bf[nt] = make16(*(const v8h*)&Bs[r][8 * g], *(const v8h*)&Bs[r][16 + 8 * g]);
    }
    for (int mt = 0; mt < 2; mt++)
      for (int nt = 0; nt < 4; nt++)
        acc[mt][nt] = __builtin_amdgcn_wmma_f32_16x16x32_f16(
            false, af[mt], false, bf[nt], (short)0, acc[mt][nt], false, false);
    __syncthreads();
  }

  for (int mt = 0; mt < 2; mt++)
    for (int nt = 0; nt < 4; nt++) {
      int col = n0 + wn * 64 + nt * 16 + n16;
      float bv = bias[col];
      for (int j = 0; j < 8; j++) {            // C/D frag: row = 8g+j, col = lane&15
        int row = m0 + wm * 32 + mt * 16 + 8 * g + j;
        float v = acc[mt][nt][j] + bv;
        if (OUT_F32) ((float*)outp)[(size_t)row * N + col] = v;
        else         ((h16*)outp)[(size_t)row * N + col]   = (h16)v;
      }
    }
}

// ---------------------------------------------------------------- causal flash attention
// grid: B*NH*(T/128) workgroups of 256 threads; wave w owns q rows [qbase+16w, +16)
__global__ __launch_bounds__(256)
void attn_kernel(const h16* __restrict__ qkv, h16* __restrict__ y) {
  __shared__ h16 Ks[32][72];        // K tile, row-major [key][dim]
  __shared__ h16 VsT[64][40];       // V tile transposed [dim][key]
  __shared__ h16 Ps[8][16][40];     // per-wave P transpose buffer [qrow][key]

  const int tid  = threadIdx.x;
  const int lane = tid & 31, w = tid >> 5;
  const int g = lane >> 4, n16 = lane & 15;
  const int nqblk = T_ / 128;
  const int qblk = blockIdx.x % nqblk;
  const int bh   = blockIdx.x / nqblk;
  const int b = bh / NH_, h = bh % NH_;
  const int qbase = qblk * 128;
  const int qrow0 = qbase + w * 16;

  // Q fragments (A-matrix, 16x32 each, 2 K-steps covering HS=64), held in regs
  const h16* qg = qkv + (size_t)(b * T_ + qrow0 + n16) * (3 * C_) + (size_t)h * HS_;
  v16h qf[2];
  for (int ks = 0; ks < 2; ks++)
    qf[ks] = make16(*(const v8h*)(qg + ks * 32 + 8 * g),
                    *(const v8h*)(qg + ks * 32 + 16 + 8 * g));

  v8f o[4]; for (int d = 0; d < 4; d++) o[d] = {};
  float mrow[8], lrow[8];
  for (int j = 0; j < 8; j++) { mrow[j] = -3.0e38f; lrow[j] = 0.0f; }

  const int key_s = tid >> 3, c0 = (tid & 7) * 8;     // staging coords
  const unsigned ldsK = lds_off(&Ks[key_s][c0]);

  const int nkb = (qbase + 128) / 32;          // all waves run same count (masked)
  for (int kb = 0; kb < nkb; kb++) {
    const int kstart = kb * 32;
    {                                          // stage K (async) and V^T (transposing)
      const h16* kg = qkv + (size_t)(b * T_ + kstart + key_s) * (3 * C_) + C_ + h * HS_ + c0;
      async_cp16(ldsK, kg);                    // global_load_async_to_lds_b128
      const h16* vg = qkv + (size_t)(b * T_ + kstart + key_s) * (3 * C_) + 2 * C_ + h * HS_ + c0;
      v8h vv = *(const v8h*)vg;
      for (int i = 0; i < 8; i++) VsT[c0 + i][key_s] = vv[i];
    }
    wait_async0();
    __syncthreads();

    // S = Q*K^T : two 16x16 key sub-tiles, 2 K-steps each
    v8f s0 = {}, s1 = {};
    for (int ks = 0; ks < 2; ks++) {
      v16h b0 = make16(*(const v8h*)&Ks[n16][ks * 32 + 8 * g],
                       *(const v8h*)&Ks[n16][ks * 32 + 16 + 8 * g]);
      v16h b1 = make16(*(const v8h*)&Ks[16 + n16][ks * 32 + 8 * g],
                       *(const v8h*)&Ks[16 + n16][ks * 32 + 16 + 8 * g]);
      s0 = __builtin_amdgcn_wmma_f32_16x16x32_f16(false, qf[ks], false, b0, (short)0, s0, false, false);
      s1 = __builtin_amdgcn_wmma_f32_16x16x32_f16(false, qf[ks], false, b1, (short)0, s1, false, false);
    }

    // scale 1/sqrt(64), causal mask, online softmax (rows live across 16 lanes)
    float p0[8], p1[8], alpha[8];
    for (int j = 0; j < 8; j++) {
      int qr = qbase + w * 16 + 8 * g + j;
      float a = s0[j] * 0.125f;
      float c = s1[j] * 0.125f;
      if (kstart + n16 > qr)      a = -3.0e38f;
      if (kstart + 16 + n16 > qr) c = -3.0e38f;
      p0[j] = a; p1[j] = c;
    }
    for (int j = 0; j < 8; j++) {
      float rm = fmaxf(p0[j], p1[j]);
      for (int msk = 1; msk < 16; msk <<= 1)
        rm = fmaxf(rm, __shfl_xor(rm, msk, 32));
      float mnew = fmaxf(mrow[j], rm);
      alpha[j] = __expf(mrow[j] - mnew);
      p0[j] = __expf(p0[j] - mnew);
      p1[j] = __expf(p1[j] - mnew);
      float rs = p0[j] + p1[j];
      for (int msk = 1; msk < 16; msk <<= 1)
        rs += __shfl_xor(rs, msk, 32);
      lrow[j] = lrow[j] * alpha[j] + rs;
      mrow[j] = mnew;
    }
    for (int d = 0; d < 4; d++)
      for (int j = 0; j < 8; j++)
        o[d][j] *= alpha[j];

    // transpose P through per-wave LDS (DS ops are in-order per wave)
    for (int j = 0; j < 8; j++) {
      Ps[w][8 * g + j][n16]      = (h16)p0[j];
      Ps[w][8 * g + j][16 + n16] = (h16)p1[j];
    }
    asm volatile("s_wait_dscnt 0x0" ::: "memory");
    v16h pf = make16(*(const v8h*)&Ps[w][n16][8 * g],
                     *(const v8h*)&Ps[w][n16][16 + 8 * g]);
    for (int d = 0; d < 4; d++) {
      v16h vf = make16(*(const v8h*)&VsT[d * 16 + n16][8 * g],
                       *(const v8h*)&VsT[d * 16 + n16][16 + 8 * g]);
      o[d] = __builtin_amdgcn_wmma_f32_16x16x32_f16(false, pf, false, vf, (short)0, o[d], false, false);
    }
    __syncthreads();
  }

  for (int j = 0; j < 8; j++) {
    float inv = 1.0f / lrow[j];
    int qr = qbase + w * 16 + 8 * g + j;
    h16* yp = y + (size_t)(b * T_ + qr) * C_ + (size_t)h * HS_;
    for (int d = 0; d < 4; d++)
      yp[d * 16 + n16] = (h16)(o[d][j] * inv);
  }
}

// ---------------------------------------------------------------- launch
extern "C" void kernel_launch(void* const* d_in, const int* in_sizes, int n_in,
                              void* d_out, int out_size, void* d_ws, size_t ws_size,
                              hipStream_t stream) {
  (void)in_sizes; (void)n_in; (void)out_size; (void)ws_size;
  const float* x      = (const float*)d_in[0];
  const float* W_attn = (const float*)d_in[1];
  const float* b_attn = (const float*)d_in[2];
  const float* W_o    = (const float*)d_in[3];
  const float* b_o    = (const float*)d_in[4];
  float* out = (float*)d_out;

  char* ws = (char*)d_ws;
  h16* x_h   = (h16*)(ws);                           // 16 MB
  h16* Wat_h = (h16*)(ws + 16777216);                //  6 MB
  h16* Wo_h  = (h16*)(ws + 23068672);                //  2 MB
  h16* qkv_h = (h16*)(ws + 25165824);                // 48 MB
  h16* y_h   = (h16*)(ws + 75497472);                // 16 MB  (total ~88 MB)

  cvt_f32_f16<<<2048, 256, 0, stream>>>(x,      x_h,   B_ * T_ * C_);
  cvt_f32_f16<<<2048, 256, 0, stream>>>(W_attn, Wat_h, 3 * C_ * C_);
  cvt_f32_f16<<<1024, 256, 0, stream>>>(W_o,    Wo_h,  C_ * C_);

  gemm_bias<false><<<dim3((3 * C_) / 128, (B_ * T_) / 128), 256, 0, stream>>>(
      x_h, Wat_h, b_attn, qkv_h, B_ * T_, 3 * C_, C_);

  rope_kernel<<<(B_ * T_ * NH_ + 255) / 256, 256, 0, stream>>>(qkv_h);

  attn_kernel<<<B_ * NH_ * (T_ / 128), 256, 0, stream>>>(qkv_h, y_h);

  gemm_bias<true><<<dim3(C_ / 128, (B_ * T_) / 128), 256, 0, stream>>>(
      y_h, Wo_h, b_o, out, B_ * T_, C_, C_);
}
